// BitNetLinear_16956530884783
// MI455X (gfx1250) — matrix operand order repaired and checked by reading, not compiled, not run on hardware
//
#include <hip/hip_runtime.h>

typedef __attribute__((ext_vector_type(16))) __bf16 v16bf;
typedef __attribute__((ext_vector_type(8)))  float  v8f;
typedef __attribute__((ext_vector_type(4)))  float  v4f;

#define BATCH   32
#define IN_DIM  4096
#define OUT_DIM 11008
#define NSPLIT  16
#define KCHUNKS (IN_DIM / 32)                  // 128 K-chunks of 32
#define FRAG_ELEMS 512                         // 32 lanes * 16 bf16 per fragment
#define FRAG_TOTAL_BYTES (KCHUNKS * 4 * FRAG_ELEMS * 2)  // chunks*(2 Mtiles x hi/lo)*1KB = 512KB

// Load 8 consecutive fp32, split each into bf16 hi + bf16 lo (residual),
// place into fragment positions [base, base+8).
__device__ __forceinline__ void load_split8(const float* __restrict__ p,
                                            v16bf& hi, v16bf& lo, int base) {
  v4f f0 = *reinterpret_cast<const v4f*>(p);
  v4f f1 = *reinterpret_cast<const v4f*>(p + 4);
#pragma unroll
  for (int j = 0; j < 4; ++j) {
    float f  = f0[j];
    __bf16 h = (__bf16)f;
    hi[base + j] = h;
    lo[base + j] = (__bf16)(f - (float)h);
  }
#pragma unroll
  for (int j = 0; j < 4; ++j) {
    float f  = f1[j];
    __bf16 h = (__bf16)f;
    hi[base + 4 + j] = h;
    lo[base + 4 + j] = (__bf16)(f - (float)h);
  }
}

// Build a WMMA-A bf16 fragment (ISA 16-bit A 16x32 layout) for X rows
// [tile*16, tile*16+16), K chunk c. lanes 0-15: K = {0..7,16..23}; lanes
// 16-31: K = {8..15,24..31}.
__device__ __forceinline__ void make_a_frag(const float* __restrict__ X,
                                            int c, int tile, int laneM, int laneSel,
                                            v16bf& hi, v16bf& lo) {
  const float* xrow = X + (tile * 16 + laneM) * IN_DIM;
  const int ka = c * 32 + 8 * laneSel;
  load_split8(xrow + ka,      hi, lo, 0);
  load_split8(xrow + ka + 16, hi, lo, 8);
}

// Phase 0: pre-split X into A-fragments in ws. Layout (element units):
// frag[((c*2 + tile)*2 + part) * 512 + lane*16], part: 0=hi, 1=lo.
__global__ __launch_bounds__(256) void xfrag_kernel(const float* __restrict__ X,
                                                    __bf16* __restrict__ frag) {
  const int lane    = threadIdx.x & 31;
  const int laneM   = lane & 15;
  const int laneSel = lane >> 4;
  const int idx  = blockIdx.x * (blockDim.x >> 5) + (threadIdx.x >> 5); // 0..255
  const int c    = idx >> 1;
  const int tile = idx & 1;
  if (c >= KCHUNKS) return;

  v16bf hi, lo;
  make_a_frag(X, c, tile, laneM, laneSel, hi, lo);

  __bf16* hp = frag + (size_t)((c * 2 + tile) * 2 + 0) * FRAG_ELEMS + lane * 16;
  __bf16* lp = frag + (size_t)((c * 2 + tile) * 2 + 1) * FRAG_ELEMS + lane * 16;
  *reinterpret_cast<v16bf*>(hp) = hi;
  *reinterpret_cast<v16bf*>(lp) = lo;
}

// Convert fp32 ternary weights wp[0..8) and wp[16..24) to a bf16 B fragment row-chunk.
__device__ __forceinline__ v16bf make_b_frag(const float* __restrict__ wp) {
  v16bf b;
#pragma unroll
  for (int c4 = 0; c4 < 4; ++c4) {
    v4f f = *reinterpret_cast<const v4f*>(wp + 4 * c4 + (c4 >= 2 ? 8 : 0));
#pragma unroll
    for (int j = 0; j < 4; ++j) b[4 * c4 + j] = (__bf16)f[j];
  }
  return b;
}

// Main GEMM: one wave owns TWO 16-wide N tiles x both M tiles over [kBegin,kEnd).
// blockIdx.y = K-split. USE_FRAG: read pre-split A fragments from ws (hot path).
template <bool USE_FRAG>
__global__ __launch_bounds__(256) void bitnet_wmma_kernel(
    const float* __restrict__ X, const float* __restrict__ W,
    const float* __restrict__ scale_p, const __bf16* __restrict__ frag,
    float* __restrict__ out, int kChunk, int applyScale) {
  const int lane    = threadIdx.x & 31;   // wave32
  const int laneM   = lane & 15;
  const int laneSel = lane >> 4;

  const int pair  = blockIdx.x * (blockDim.x >> 5) + (threadIdx.x >> 5);
  const int nBase = pair * 32;
  if (nBase >= OUT_DIM) return;           // wave-uniform (grid is exact anyway)

  const int kBegin = blockIdx.y * kChunk;
  const int kEnd   = kBegin + kChunk;

  const float* wrow0 = W + (size_t)(nBase + laneM) * IN_DIM;       // N tile 0
  const float* wrow1 = W + (size_t)(nBase + 16 + laneM) * IN_DIM;  // N tile 1
  const __bf16* fbase = frag + lane * 16;                          // per-lane slice

  v8f acc00 = {}, acc01 = {};  // N tile 0: M 0..15 / 16..31
  v8f acc10 = {}, acc11 = {};  // N tile 1

  for (int k0 = kBegin; k0 < kEnd; k0 += 32) {
    const int c = k0 >> 5;
    __builtin_prefetch(wrow0 + k0 + 256, 0, 1);   // global_prefetch_b8
    __builtin_prefetch(wrow1 + k0 + 256, 0, 1);

    // ---- A fragments (pre-split bf16 hi/lo, 4KB/chunk, L2-resident)
    v16bf a0h, a0l, a1h, a1l;
    if (USE_FRAG) {
      const __bf16* f = fbase + (size_t)c * (4 * FRAG_ELEMS);
      a0h = *reinterpret_cast<const v16bf*>(f);
      a0l = *reinterpret_cast<const v16bf*>(f + FRAG_ELEMS);
      a1h = *reinterpret_cast<const v16bf*>(f + 2 * FRAG_ELEMS);
      a1l = *reinterpret_cast<const v16bf*>(f + 3 * FRAG_ELEMS);
    } else {
      make_a_frag(X, c, 0, laneM, laneSel, a0h, a0l);
      make_a_frag(X, c, 1, laneM, laneSel, a1h, a1l);
    }

    // ---- B fragments (32x16 KxN): lane = column, lanes 0-15 K={0..7,16..23},
    // lanes 16-31 K={8..15,24..31}. Ternary weights exact in bf16.
    v16bf b0 = make_b_frag(wrow0 + k0 + 8 * laneSel);
    v16bf b1 = make_b_frag(wrow1 + k0 + 8 * laneSel);

    acc00 = __builtin_amdgcn_wmma_f32_16x16x32_bf16(false, a0h, false, b0, (short)0, acc00, false, false);
    acc00 = __builtin_amdgcn_wmma_f32_16x16x32_bf16(false, a0l, false, b0, (short)0, acc00, false, false);
    acc01 = __builtin_amdgcn_wmma_f32_16x16x32_bf16(false, a1h, false, b0, (short)0, acc01, false, false);
    acc01 = __builtin_amdgcn_wmma_f32_16x16x32_bf16(false, a1l, false, b0, (short)0, acc01, false, false);
    acc10 = __builtin_amdgcn_wmma_f32_16x16x32_bf16(false, a0h, false, b1, (short)0, acc10, false, false);
    acc10 = __builtin_amdgcn_wmma_f32_16x16x32_bf16(false, a0l, false, b1, (short)0, acc10, false, false);
    acc11 = __builtin_amdgcn_wmma_f32_16x16x32_bf16(false, a1h, false, b1, (short)0, acc11, false, false);
    acc11 = __builtin_amdgcn_wmma_f32_16x16x32_bf16(false, a1l, false, b1, (short)0, acc11, false, false);
  }

  // ---- store C: N = nBase+laneM (+16 for tile 1); vgpr r -> M = 8*laneSel + r
  const float scale = applyScale ? scale_p[0] : 1.0f;
  float* dst = out + (size_t)blockIdx.y * (BATCH * OUT_DIM);
  const int n0 = nBase + laneM;
#pragma unroll
  for (int r = 0; r < 8; ++r) {
    const size_t m0 = (size_t)(8 * laneSel + r) * OUT_DIM;
    const size_t m1 = (size_t)(16 + 8 * laneSel + r) * OUT_DIM;
    float v00 = acc00[r], v01 = acc01[r], v10 = acc10[r], v11 = acc11[r];
    if (applyScale) { v00 *= scale; v01 *= scale; v10 *= scale; v11 *= scale; }
    dst[m0 + n0]      = v00;
    dst[m1 + n0]      = v01;
    dst[m0 + n0 + 16] = v10;
    dst[m1 + n0 + 16] = v11;
  }
}

// Phase 2: deterministic fixed-order reduction of K-split partials + scale.
__global__ __launch_bounds__(256) void bitnet_reduce_kernel(
    const float* __restrict__ part, const float* __restrict__ scale_p,
    float* __restrict__ Y, int S) {
  const int i = blockIdx.x * blockDim.x + threadIdx.x;
  if (i >= BATCH * OUT_DIM) return;
  float sum = 0.0f;
  for (int s = 0; s < S; ++s) sum += part[(size_t)s * (BATCH * OUT_DIM) + i];
  Y[i] = sum * scale_p[0];
}

extern "C" void kernel_launch(void* const* d_in, const int* in_sizes, int n_in,
                              void* d_out, int out_size, void* d_ws, size_t ws_size,
                              hipStream_t stream) {
  const float* x = (const float*)d_in[0];   // [32,4096] f32
  const float* w = (const float*)d_in[1];   // [11008,4096] f32 ternary
  const float* s = (const float*)d_in[2];   // [1] f32 scale
  float* y = (float*)d_out;                 // [32,11008] f32

  const int pairs = OUT_DIM / 32;           // 344 (two 16-wide tiles per wave)
  const int wavesPerBlock = 8;
  dim3 block(32 * wavesPerBlock);
  const int gx = (pairs + wavesPerBlock - 1) / wavesPerBlock;  // 43

  const size_t partialBytes = (size_t)NSPLIT * BATCH * OUT_DIM * sizeof(float); // 22.5 MB
  const size_t fragBytes    = FRAG_TOTAL_BYTES;                                 // 512 KB
  const int elems = BATCH * OUT_DIM;

  if (d_ws != nullptr && ws_size >= partialBytes + fragBytes) {
    // Hot path: pre-split X, split-K=16 (5504 waves), deterministic reduce.
    __bf16* frag = (__bf16*)((char*)d_ws + partialBytes);
    xfrag_kernel<<<dim3(KCHUNKS * 2 / wavesPerBlock), block, 0, stream>>>(x, frag);
    bitnet_wmma_kernel<true><<<dim3(gx, NSPLIT), block, 0, stream>>>(
        x, w, s, frag, (float*)d_ws, IN_DIM / NSPLIT, /*applyScale=*/0);
    bitnet_reduce_kernel<<<(elems + 255) / 256, 256, 0, stream>>>(
        (const float*)d_ws, s, y, NSPLIT);
  } else if (d_ws != nullptr && ws_size >= fragBytes) {
    __bf16* frag = (__bf16*)d_ws;
    xfrag_kernel<<<dim3(KCHUNKS * 2 / wavesPerBlock), block, 0, stream>>>(x, frag);
    bitnet_wmma_kernel<true><<<dim3(gx, 1), block, 0, stream>>>(
        x, w, s, frag, y, IN_DIM, /*applyScale=*/1);
  } else {
    // No usable scratch: inline conversion, single split.
    bitnet_wmma_kernel<false><<<dim3(gx, 1), block, 0, stream>>>(
        x, w, s, nullptr, y, IN_DIM, /*applyScale=*/1);
  }
}
